// FourierModule_5205500362753
// MI455X (gfx1250) — matrix operand order repaired and checked by reading, not compiled, not run on hardware
//
#include <hip/hip_runtime.h>

typedef unsigned short u16;
typedef unsigned int   u32;
typedef __attribute__((ext_vector_type(16))) __bf16 v16bf;
typedef __attribute__((ext_vector_type(8)))  float  v8f;

#define ALPHA 0.3f
#define EPS   1e-3f
#define NBINS 4096
#define BATCH 16
#define HH 128
#define WW 128
#define CC 64
#define PIX (BATCH*HH*WW)          // 262144 pixels
#define ELEMS (PIX*CC)             // 16777216 elements
#define PER_B (HH*WW*CC)           // 1<<20

// ---------------------------------------------------------------- utilities
__device__ __forceinline__ float leaky(float x) { return x >= 0.f ? x : ALPHA * x; }

__device__ __forceinline__ u16 f2bf(float f) {
    union { float f; u32 u; } v; v.f = f;
    u32 r = v.u + 0x7fffu + ((v.u >> 16) & 1u);   // round-to-nearest-even
    return (u16)(r >> 16);
}
__device__ __forceinline__ u32 fenc(float f) {    // order-preserving float->uint
    u32 u = __float_as_uint(f);
    return (u & 0x80000000u) ? ~u : (u | 0x80000000u);
}
__device__ __forceinline__ float fdec(u32 e) {
    u32 u = (e & 0x80000000u) ? (e & 0x7fffffffu) : ~e;
    return __uint_as_float(u);
}

__global__ void fill32_kernel(u32* p, u32 val, int n) {
    int i = blockIdx.x * blockDim.x + threadIdx.x;
    for (; i < n; i += gridDim.x * blockDim.x) p[i] = val;
}

// ---------------------------------------------------------------- FFT pass
// Batched radix-2 DIT FFT along a strided axis, one line in LDS.
// base(line) = (line / innerCount) * outerStride + (line % innerCount)
template<int LEN, int LOG2>
__global__ void fft_pass(const float2* __restrict__ cin,
                         const float*  __restrict__ rin,
                         float2* __restrict__ cout,
                         float*  __restrict__ rout,
                         int nLines, int stride, int innerCount, int outerStride,
                         float dirSign, float scale)
{
    constexpr int HALF  = LEN / 2;
    constexpr int LINES = 256 / HALF;
    __shared__ float2 tile[LINES][LEN];

    const int tid  = threadIdx.x;
    const int ll   = tid / HALF;
    const int t    = tid % HALF;
    const int line = blockIdx.x * LINES + ll;
    const bool act = line < nLines;
    int base = 0;
    if (act) base = (line / innerCount) * outerStride + (line % innerCount);

    // load bit-reversed
    #pragma unroll
    for (int e = t; e < LEN; e += HALF) {
        int r = (int)(__brev((u32)e) >> (32 - LOG2));
        float2 v = make_float2(0.f, 0.f);
        if (act) {
            int addr = base + e * stride;
            if (rin) v.x = rin[addr];
            else     v   = cin[addr];
        }
        tile[ll][r] = v;
    }
    // stages
    #pragma unroll
    for (int s = 1; s <= LOG2; ++s) {
        __syncthreads();
        int m = 1 << s, half = m >> 1;
        int j  = t & (half - 1);
        int i0 = (t / half) * m + j;
        int i1 = i0 + half;
        float ang = dirSign * 6.28318530717958f * (float)j / (float)m;
        float sw, cw; __sincosf(ang, &sw, &cw);
        float2 a = tile[ll][i0], b = tile[ll][i1];
        float2 tw = make_float2(cw * b.x - sw * b.y, cw * b.y + sw * b.x);
        tile[ll][i0] = make_float2(a.x + tw.x, a.y + tw.y);
        tile[ll][i1] = make_float2(a.x - tw.x, a.y - tw.y);
    }
    __syncthreads();
    if (act) {
        #pragma unroll
        for (int e = t; e < LEN; e += HALF) {
            int addr = base + e * stride;
            float2 v = tile[ll][e];
            if (rout) rout[addr] = v.x * scale;
            else      cout[addr] = make_float2(v.x * scale, v.y * scale);
        }
    }
}

// ---------------------------------------------------------------- batch statistics
__global__ void batch_stats_kernel(const float2* __restrict__ F,
                                   float* bsum, float* bsumsq, u32* bminE, u32* bmaxE)
{
    const int BPB = 64;
    int b   = blockIdx.x / BPB;
    int blk = blockIdx.x % BPB;
    int tid = threadIdx.x;
    const float2* Fb = F + (size_t)b * PER_B;
    float s = 0.f, s2 = 0.f, mn = 3.4e38f, mx = -3.4e38f;
    for (int i = blk * 256 + tid; i < PER_B; i += BPB * 256) {
        float v = Fb[i].x;
        s += v; s2 += v * v; mn = fminf(mn, v); mx = fmaxf(mx, v);
    }
    __shared__ float ls[256], ls2[256], lmn[256], lmx[256];
    ls[tid] = s; ls2[tid] = s2; lmn[tid] = mn; lmx[tid] = mx;
    for (int off = 128; off > 0; off >>= 1) {
        __syncthreads();
        if (tid < off) {
            ls[tid]  += ls[tid + off];
            ls2[tid] += ls2[tid + off];
            lmn[tid] = fminf(lmn[tid], lmn[tid + off]);
            lmx[tid] = fmaxf(lmx[tid], lmx[tid + off]);
        }
    }
    if (tid == 0) {
        atomicAdd(&bsum[b], ls[0]);
        atomicAdd(&bsumsq[b], ls2[0]);
        atomicMin(&bminE[b], fenc(lmn[0]));
        atomicMax(&bmaxE[b], fenc(lmx[0]));
    }
}

__global__ void batch_hist_kernel(const float2* __restrict__ F,
                                  const u32* bminE, const u32* bmaxE, u32* hist)
{
    const int BPB = 64;
    int b   = blockIdx.x / BPB;
    int blk = blockIdx.x % BPB;
    int tid = threadIdx.x;
    __shared__ u32 lh[NBINS];
    for (int i = tid; i < NBINS; i += 256) lh[i] = 0u;
    __syncthreads();
    float mn = fdec(bminE[b]), mx = fdec(bmaxE[b]);
    float invw = (float)NBINS / fmaxf(mx - mn, 1e-30f);
    const float2* Fb = F + (size_t)b * PER_B;
    for (int i = blk * 256 + tid; i < PER_B; i += BPB * 256) {
        float v = Fb[i].x;
        int bin = (int)((v - mn) * invw);
        bin = min(max(bin, 0), NBINS - 1);
        atomicAdd(&lh[bin], 1u);
    }
    __syncthreads();
    for (int i = tid; i < NBINS; i += 256)
        if (lh[i]) atomicAdd(&hist[b * NBINS + i], lh[i]);
}

__global__ void percentile_kernel(const u32* __restrict__ hist,
                                  const float* bsum, const float* bsumsq,
                                  const u32* bminE, const u32* bmaxE,
                                  float* statsFlat)   // [21][16] row-major
{
    int b = blockIdx.x;
    __shared__ float cdf[NBINS];
    __shared__ u32 h[NBINS];
    for (int i = threadIdx.x; i < NBINS; i += 256) h[i] = hist[b * NBINS + i];
    __syncthreads();
    if (threadIdx.x == 0) {
        float c = 0.f;
        for (int i = 0; i < NBINS; ++i) { c += (float)h[i]; cdf[i] = c; }
    }
    __syncthreads();
    const float n  = (float)PER_B;
    float mn = fdec(bminE[b]), mx = fdec(bmaxE[b]);
    float w  = (mx - mn) / (float)NBINS;
    if (threadIdx.x < 19) {
        int qi = threadIdx.x;
        float q = 5.f * (float)(qi + 1);
        float rank = q * 0.01f * (n - 1.f);
        int lo = 0, hi = NBINS - 1;
        while (lo < hi) { int mid = (lo + hi) >> 1; if (cdf[mid] > rank) hi = mid; else lo = mid + 1; }
        float cb   = (lo == 0) ? 0.f : cdf[lo - 1];
        float cnt  = fmaxf(cdf[lo] - cb, 1.f);
        float frac = (rank - cb) / cnt;
        statsFlat[(2 + qi) * 16 + b] = mn + ((float)lo + frac) * w;
    }
    if (threadIdx.x == 19) {
        float m = bsum[b] / n;
        statsFlat[0 * 16 + b] = m;
        statsFlat[1 * 16 + b] = sqrtf(fmaxf(bsumsq[b] / n - m * m, 0.f));
    }
}

// ---------------------------------------------------------------- tiny MLP (1 block, 64 threads)
__global__ void mlp_kernel(const float* __restrict__ statsFlat,
                           const float* W_in, const float* b_in,
                           const float* ln_g, const float* ln_b,
                           const float* resW, const float* resB,
                           const float* resG, const float* resBeta,
                           const float* W_out, const float* b_out,
                           float* TU, float* TL)
{
    __shared__ float hbuf[16][64], ybuf[16][64], mS[16], invS[16];
    int u = threadIdx.x;                       // 0..63

    // input dense: feat[16,21] @ W_in[21,64]
    for (int i = 0; i < 16; ++i) {
        float acc = b_in[u];
        for (int k = 0; k < 21; ++k) acc += statsFlat[i * 21 + k] * W_in[k * 64 + u];
        ybuf[i][u] = acc;
    }
    __syncthreads();
    if (u < 16) {
        float s = 0.f;
        for (int k = 0; k < 64; ++k) s += ybuf[u][k];
        float m = s * (1.f / 64.f), v = 0.f;
        for (int k = 0; k < 64; ++k) { float d = ybuf[u][k] - m; v += d * d; }
        mS[u] = m; invS[u] = rsqrtf(v * (1.f / 64.f) + EPS);
    }
    __syncthreads();
    for (int i = 0; i < 16; ++i)
        hbuf[i][u] = leaky((ybuf[i][u] - mS[i]) * invS[i] * ln_g[u] + ln_b[u]);

    // 3 residual blocks
    for (int l = 0; l < 3; ++l) {
        __syncthreads();
        for (int i = 0; i < 16; ++i) {
            float acc = resB[l * 64 + u];
            for (int k = 0; k < 64; ++k) acc += hbuf[i][k] * resW[(l * 64 + k) * 64 + u];
            ybuf[i][u] = acc;
        }
        __syncthreads();
        if (u < 16) {
            float s = 0.f;
            for (int k = 0; k < 64; ++k) s += ybuf[u][k];
            float m = s * (1.f / 64.f), v = 0.f;
            for (int k = 0; k < 64; ++k) { float d = ybuf[u][k] - m; v += d * d; }
            mS[u] = m; invS[u] = rsqrtf(v * (1.f / 64.f) + EPS);
        }
        __syncthreads();
        for (int i = 0; i < 16; ++i)
            hbuf[i][u] += leaky((ybuf[i][u] - mS[i]) * invS[i] * resG[l * 64 + u] + resBeta[l * 64 + u]);
    }
    __syncthreads();
    if (u < 16) {
        int i = u;
        float o0 = b_out[0], o1 = b_out[1];
        for (int k = 0; k < 64; ++k) {
            o0 += hbuf[i][k] * W_out[k * 8 + 0];
            o1 += hbuf[i][k] * W_out[k * 8 + 1];
        }
        TU[i] = leaky(o0);
        TL[i] = leaky(o1);
    }
}

// ---------------------------------------------------------------- spectral filter (in-place)
__global__ void filter_kernel(float2* F, const float* __restrict__ TU, const float* __restrict__ TL)
{
    for (int i = blockIdx.x * blockDim.x + threadIdx.x; i < ELEMS; i += gridDim.x * blockDim.x) {
        int b = i >> 20;
        float tu = TU[b], tl = TL[b];
        float2 f = F[i];
        float R = f.x, I = f.y;
        float tu_f = R * fmaxf(tu - R, 0.f) / (tu * 0.1f - R + 1.f);
        float tl_f = tu_f * fmaxf(tu_f - tl, 0.f) / (tu_f - tl * 0.1f + 1.f);
        F[i] = make_float2(tl_f, I);
    }
}

// ---------------------------------------------------------------- bf16 conversion / weight packing
__global__ void f32_to_bf16_kernel(const float* __restrict__ in, u16* __restrict__ out, int n)
{
    for (int i = blockIdx.x * blockDim.x + threadIdx.x; i < n; i += gridDim.x * blockDim.x)
        out[i] = f2bf(in[i]);
}

// Pack W[Ktot][64] fp32 into per-lane WMMA B fragments:
// Bp[((kc*4 + nt)*32 + lane)*16 + h] = bf16( W[kc*32 + (h<8 ? (lane/16)*8 + h : 16 + (lane/16)*8 + h-8)][nt*16 + lane%16] )
__global__ void pack_weights_kernel(const float* __restrict__ Wsrc, u16* __restrict__ Bp, int Ktot)
{
    int KC = Ktot / 32;
    int total = KC * 4 * 32 * 16;
    int idx = blockIdx.x * blockDim.x + threadIdx.x;
    if (idx >= total) return;
    int hh   = idx & 15;
    int lane = (idx >> 4) & 31;
    int nt   = (idx >> 9) & 3;
    int kc   = idx >> 11;
    int N  = nt * 16 + (lane & 15);
    int kl = (hh < 8) ? (((lane >> 4) << 3) + hh) : (16 + ((lane >> 4) << 3) + (hh - 8));
    int k  = kc * 32 + kl;
    Bp[idx] = f2bf(Wsrc[k * 64 + N]);
}

// ---------------------------------------------------------------- WMMA fragment loads
__device__ __forceinline__ v16bf load_afrag(const u16* p)   // halves 0..7 = K..K+7, 8..15 = K+16..K+23
{
    union { uint4 q; u16 u[8]; } a, b;
    a.q = *(const uint4*)(p);
    b.q = *(const uint4*)(p + 16);
    union { v16bf v; u16 u[16]; } r;
    #pragma unroll
    for (int j = 0; j < 8; ++j) { r.u[j] = a.u[j]; r.u[8 + j] = b.u[j]; }
    return r.v;
}
__device__ __forceinline__ v16bf load_bfrag(const u16* p)   // contiguous pre-packed 32B
{
    union { uint4 q[2]; v16bf v; } r;
    r.q[0] = *(const uint4*)(p);
    r.q[1] = *(const uint4*)(p + 8);
    return r.v;
}
__device__ __forceinline__ v16bf zero_frag()
{
    union { v16bf v; u16 u[16]; } r;
    #pragma unroll
    for (int j = 0; j < 16; ++j) r.u[j] = 0;
    return r.v;
}

// ---------------------------------------------------------------- implicit-GEMM conv via WMMA bf16
// TAPS=1: 1x1 conv (K=64).  TAPS=9: 3x3 SAME conv (K=576).
// A: [PIX][64] bf16 activations. Bp: packed weights. Out: [PIX][64] fp32 (+bias).
template<int TAPS>
__global__ void __launch_bounds__(256)
conv_wmma_kernel(const u16* __restrict__ A, const u16* __restrict__ Bp,
                 const float* __restrict__ bias, float* __restrict__ Out)
{
    const int wave = threadIdx.x >> 5;
    const int lane = threadIdx.x & 31;
    const int tileM = blockIdx.x * 8 + wave;
    const int p0 = tileM * 16;
    const int Mr = lane & 15;
    const int Kb = (lane >> 4) * 8;
    const int p  = p0 + Mr;
    const int w  = p & (WW - 1);
    const int h  = (p >> 7) & (HH - 1);

    v8f acc[4];
    #pragma unroll
    for (int nt = 0; nt < 4; ++nt)
        #pragma unroll
        for (int r = 0; r < 8; ++r) acc[nt][r] = 0.f;

    #pragma unroll
    for (int kc = 0; kc < TAPS * 2; ++kc) {
        const int tap   = kc >> 1;
        const int cBase = (kc & 1) * 32;
        const int dy = (TAPS == 1) ? 0 : (tap / 3 - 1);
        const int dx = (TAPS == 1) ? 0 : (tap % 3 - 1);
        v16bf afrag;
        if (TAPS == 1 || ((unsigned)(h + dy) < (unsigned)HH && (unsigned)(w + dx) < (unsigned)WW)) {
            int np = p + dy * WW + dx;
            afrag = load_afrag(A + (size_t)np * 64 + cBase + Kb);
        } else {
            afrag = zero_frag();
        }
        const u16* bbase = Bp + ((size_t)(kc * 4) * 32 + lane) * 16;
        #pragma unroll
        for (int nt = 0; nt < 4; ++nt) {
            v16bf bfrag = load_bfrag(bbase + (size_t)nt * 32 * 16);
            acc[nt] = __builtin_amdgcn_wmma_f32_16x16x32_bf16(
                false, afrag, false, bfrag, (short)0, acc[nt], false, false);
        }
    }
    // D layout: lane holds N = lane%16; VGPR r holds M = r + 8*(lane/16)
    const int hiM = (lane >> 4) * 8;
    #pragma unroll
    for (int nt = 0; nt < 4; ++nt) {
        int n = nt * 16 + (lane & 15);
        float bv = bias[n];
        #pragma unroll
        for (int r = 0; r < 8; ++r) {
            int pr = p0 + hiM + r;
            Out[(size_t)pr * 64 + n] = acc[nt][r] + bv;
        }
    }
}

// ---------------------------------------------------------------- per-channel (BN) stats
__global__ void channel_stats_kernel(const float* __restrict__ in, float* chs, float* chs2, int rows)
{
    int tid = threadIdx.x;
    int c   = tid & 63;
    int rl  = tid >> 6;                 // 0..3
    float s = 0.f, s2 = 0.f;
    for (int r = blockIdx.x * 4 + rl; r < rows; r += gridDim.x * 4) {
        float v = in[(size_t)r * 64 + c];
        s += v; s2 += v * v;
    }
    __shared__ float ls[256], ls2[256];
    ls[tid] = s; ls2[tid] = s2;
    __syncthreads();
    if (tid < 64) {
        float a  = ls[tid]  + ls[tid + 64]  + ls[tid + 128]  + ls[tid + 192];
        float a2 = ls2[tid] + ls2[tid + 64] + ls2[tid + 128] + ls2[tid + 192];
        atomicAdd(&chs[tid], a);
        atomicAdd(&chs2[tid], a2);
    }
}

// final = x + 4 * leaky(BN(conv1x1_out))
__global__ void cbl_post_kernel(const float* __restrict__ y, const float* __restrict__ x,
                                const float* __restrict__ chs, const float* __restrict__ chs2,
                                const float* __restrict__ g, const float* __restrict__ bta,
                                float* __restrict__ outF)
{
    const float invN = 1.f / (float)PIX;
    for (int i = blockIdx.x * blockDim.x + threadIdx.x; i < ELEMS; i += gridDim.x * blockDim.x) {
        int c = i & 63;
        float m = chs[c] * invN;
        float var = fmaxf(chs2[c] * invN - m * m, 0.f);
        float t = (y[i] - m) * rsqrtf(var + EPS) * g[c] + bta[c];
        outF[i] = x[i] + 4.f * leaky(t);
    }
}

// bf16(BN2(final)) — input to the 3x3 conv
__global__ void bn_to_bf16_kernel(const float* __restrict__ in,
                                  const float* __restrict__ chs, const float* __restrict__ chs2,
                                  const float* __restrict__ g, const float* __restrict__ bta,
                                  u16* __restrict__ out)
{
    const float invN = 1.f / (float)PIX;
    for (int i = blockIdx.x * blockDim.x + threadIdx.x; i < ELEMS; i += gridDim.x * blockDim.x) {
        int c = i & 63;
        float m = chs[c] * invN;
        float var = fmaxf(chs2[c] * invN - m * m, 0.f);
        out[i] = f2bf((in[i] - m) * rsqrtf(var + EPS) * g[c] + bta[c]);
    }
}

// ---------------------------------------------------------------- launcher
extern "C" void kernel_launch(void* const* d_in, const int* in_sizes, int n_in,
                              void* d_out, int out_size, void* d_ws, size_t ws_size,
                              hipStream_t stream)
{
    (void)in_sizes; (void)n_in; (void)out_size; (void)ws_size;

    const float* x        = (const float*)d_in[0];   // [16,128,128,64]
    const float* W_in     = (const float*)d_in[1];
    const float* b_in     = (const float*)d_in[2];
    const float* ln_in_g  = (const float*)d_in[3];
    const float* ln_in_b  = (const float*)d_in[4];
    const float* res_W    = (const float*)d_in[5];
    const float* res_b    = (const float*)d_in[6];
    const float* res_g    = (const float*)d_in[7];
    const float* res_beta = (const float*)d_in[8];
    const float* W_out    = (const float*)d_in[9];
    const float* b_out    = (const float*)d_in[10];
    const float* cbl_k    = (const float*)d_in[11];  // [1,1,64,64]
    const float* cbl_b    = (const float*)d_in[12];
    const float* cbl_g    = (const float*)d_in[13];
    const float* cbl_beta = (const float*)d_in[14];
    const float* bn_g     = (const float*)d_in[15];
    const float* bn_b     = (const float*)d_in[16];
    const float* conv_k   = (const float*)d_in[17];  // [3,3,64,64]
    const float* conv_b   = (const float*)d_in[18];
    float* out = (float*)d_out;

    // ---- workspace layout
    char* ws = (char*)d_ws;
    float2* bufA = (float2*)ws;                                   // 128 MiB
    float2* bufB = (float2*)(ws + (size_t)134217728);             // 128 MiB
    u16*    Abf  = (u16*)  (ws + (size_t)268435456);              //  32 MiB
    char*   sm   = ws + (size_t)268435456 + 33554432;
    u16*   Bpack1    = (u16*)(sm + 0);          // 8 KiB
    u16*   Bpack3    = (u16*)(sm + 8192);       // 72 KiB
    float* bsum      = (float*)(sm + 81920);
    float* bsumsq    = (float*)(sm + 82176);
    u32*   bminE     = (u32*)  (sm + 82432);
    u32*   bmaxE     = (u32*)  (sm + 82688);
    u32*   hist      = (u32*)  (sm + 82944);    // 256 KiB
    float* statsFlat = (float*)(sm + 345600);
    float* TU        = (float*)(sm + 347136);
    float* TL        = (float*)(sm + 347392);
    float* chs1      = (float*)(sm + 347648);
    float* chs1b     = (float*)(sm + 347904);
    float* chs2      = (float*)(sm + 348160);
    float* chs2b     = (float*)(sm + 348416);
    // aliases (lifetimes verified: forward-FFT data dead before reuse)
    float* featReal = (float*)bufB;                               // ifft real output
    float* conv1out = (float*)((char*)bufB + (size_t)67108864);   // 1x1 conv output
    float* finalBuf = (float*)bufA;                               // x + 4*CBL

    // ---- init small buffers
    fill32_kernel<<<1, 256, 0, stream>>>((u32*)bsum,   0u, 16);
    fill32_kernel<<<1, 256, 0, stream>>>((u32*)bsumsq, 0u, 16);
    fill32_kernel<<<1, 256, 0, stream>>>(bminE, 0xFFFFFFFFu, 16);
    fill32_kernel<<<1, 256, 0, stream>>>(bmaxE, 0u, 16);
    fill32_kernel<<<256, 256, 0, stream>>>(hist, 0u, BATCH * NBINS);
    fill32_kernel<<<1, 256, 0, stream>>>((u32*)chs1,  0u, 64);
    fill32_kernel<<<1, 256, 0, stream>>>((u32*)chs1b, 0u, 64);
    fill32_kernel<<<1, 256, 0, stream>>>((u32*)chs2,  0u, 64);
    fill32_kernel<<<1, 256, 0, stream>>>((u32*)chs2b, 0u, 64);

    // ---- pack conv weights into WMMA B-fragment order
    pack_weights_kernel<<<16, 256, 0, stream>>>(cbl_k, Bpack1, 64);
    pack_weights_kernel<<<144, 256, 0, stream>>>(conv_k, Bpack3, 576);

    // ---- forward 3D FFT: C (len 64, stride 1), W (len 128, stride 64), H (len 128, stride 8192)
    fft_pass<64, 6><<<32768, 256, 0, stream>>>(nullptr, x, bufA, nullptr,
                                               PIX, 1, 1, 64, -1.f, 1.f);
    fft_pass<128, 7><<<32768, 256, 0, stream>>>(bufA, nullptr, bufB, nullptr,
                                                BATCH * HH * CC, 64, 64, 8192, -1.f, 1.f);
    fft_pass<128, 7><<<32768, 256, 0, stream>>>(bufB, nullptr, bufA, nullptr,
                                                BATCH * WW * CC, 8192, 8192, PER_B, -1.f, 1.f);

    // ---- order statistics on Re(F)
    batch_stats_kernel<<<BATCH * 64, 256, 0, stream>>>(bufA, bsum, bsumsq, bminE, bmaxE);
    batch_hist_kernel<<<BATCH * 64, 256, 0, stream>>>(bufA, bminE, bmaxE, hist);
    percentile_kernel<<<BATCH, 256, 0, stream>>>(hist, bsum, bsumsq, bminE, bmaxE, statsFlat);

    // ---- MLP -> TU, TL
    mlp_kernel<<<1, 64, 0, stream>>>(statsFlat, W_in, b_in, ln_in_g, ln_in_b,
                                     res_W, res_b, res_g, res_beta, W_out, b_out, TU, TL);

    // ---- spectral filter (in place on bufA)
    filter_kernel<<<32768, 256, 0, stream>>>(bufA, TU, TL);

    // ---- inverse 3D FFT: H, W, then C (real output), scale 1/(128*128*64)
    fft_pass<128, 7><<<32768, 256, 0, stream>>>(bufA, nullptr, bufB, nullptr,
                                                BATCH * WW * CC, 8192, 8192, PER_B, 1.f, 1.f / 128.f);
    fft_pass<128, 7><<<32768, 256, 0, stream>>>(bufB, nullptr, bufA, nullptr,
                                                BATCH * HH * CC, 64, 64, 8192, 1.f, 1.f / 128.f);
    fft_pass<64, 6><<<32768, 256, 0, stream>>>(bufA, nullptr, nullptr, featReal,
                                               PIX, 1, 1, 64, 1.f, 1.f / 64.f);

    // ---- CBL: bf16 convert -> 1x1 conv (WMMA) -> BN stats -> leaky + residual
    f32_to_bf16_kernel<<<32768, 256, 0, stream>>>(featReal, Abf, ELEMS);
    conv_wmma_kernel<1><<<2048, 256, 0, stream>>>(Abf, Bpack1, cbl_b, conv1out);
    channel_stats_kernel<<<256, 256, 0, stream>>>(conv1out, chs1, chs1b, PIX);
    cbl_post_kernel<<<32768, 256, 0, stream>>>(conv1out, x, chs1, chs1b, cbl_g, cbl_beta, finalBuf);

    // ---- BN2 -> bf16 -> 3x3 conv (WMMA) -> d_out
    channel_stats_kernel<<<256, 256, 0, stream>>>(finalBuf, chs2, chs2b, PIX);
    bn_to_bf16_kernel<<<32768, 256, 0, stream>>>(finalBuf, chs2, chs2b, bn_g, bn_b, Abf);
    conv_wmma_kernel<9><<<2048, 256, 0, stream>>>(Abf, Bpack3, conv_b, out);
}
